// CliffordLinear_38208029065323
// MI455X (gfx1250) — compile-verified, matrix operand order checked by reading, not run on hardware
//
#include <hip/hip_runtime.h>
#include <hip/hip_bf16.h>

// ---------------------------------------------------------------------------
// CliffordLinear on MI455X (gfx1250):
//   out[b, o, l] = sum_{i,k} sign(l^k,k) * W[o,i,l^k] * X[b,i,k] + bias[o,l]
// == GEMM: out[8192 x 2048] = X[8192 x 2048] * Wt^T,  Wt[n][kk] precomputed.
// Compute path: error-compensated bf16 split (hi/lo), 3x v_wmma_f32_16x16x32_bf16
// per K=32 chunk, f32 accumulation (~2^-16 relative error vs f32 reference).
// Global loads are register-pipelined one K-tile ahead to hide HBM/L2 latency
// behind the WMMA stream (single LDS buffer, two barriers per tile).
// ---------------------------------------------------------------------------

typedef __attribute__((ext_vector_type(16))) __bf16 v16bf;
typedef __attribute__((ext_vector_type(8)))  __bf16 v8bf;
typedef __attribute__((ext_vector_type(8)))  float  v8f;

#define BATCH 8192
#define KDIM  2048   // Cin * 8
#define NDIM  2048   // Cout * 8
#define BM    128
#define BN    128
#define BK    32
#define LDSS  40     // 32 + 8 pad (80B rows: multiple of 16B, conflict-free b128 frag reads)

__device__ __forceinline__ float blade_sign(int a, int b) {
    a >>= 1;
    int s = 0;
    while (a) { s += __builtin_popcount(a & b); a >>= 1; }
    return (s & 1) ? -1.0f : 1.0f;
}

// Split f32 into bf16 hi + bf16 lo (truncation split; lo captures next 8 mantissa bits)
__device__ __forceinline__ void split1(float x, unsigned short& h, unsigned short& l) {
    unsigned u  = __float_as_uint(x);
    unsigned hu = u & 0xffff0000u;
    h = (unsigned short)(hu >> 16);
    float r = x - __uint_as_float(hu);
    l = (unsigned short)(__float_as_uint(r) >> 16);
}

// ---------------------------------------------------------------------------
// Kernel 1: build Wt[n][kk] = sign(l^k,k) * W[o,i,l^k], pre-split into bf16 hi/lo.
//   n = o*8 + l  (output blade-channel),  kk = i*8 + k  (input blade-channel)
// ---------------------------------------------------------------------------
__global__ void clifford_prep_wt(const float* __restrict__ w,
                                 unsigned short* __restrict__ bh,
                                 unsigned short* __restrict__ bl) {
    int idx = blockIdx.x * 256 + threadIdx.x;     // enumerates [n][kk], kk fastest
    int n  = idx >> 11;
    int kk = idx & (KDIM - 1);
    int i = kk >> 3, k = kk & 7;
    int o = n  >> 3, l = n  & 7;
    int j = l ^ k;
    float v = blade_sign(j, k) * w[o * 2048 + i * 8 + j];
    unsigned short h, lo;
    split1(v, h, lo);
    bh[idx] = h;
    bl[idx] = lo;
}

// ---------------------------------------------------------------------------
// Kernel 2: tiled GEMM, bf16x3 compensated, WMMA 16x16x32.
// Block: 256 threads = 8 waves, wave grid 4(M) x 2(N); wave tile 32x64 (2x4 acc tiles).
// ---------------------------------------------------------------------------
__global__ __launch_bounds__(256, 1) void clifford_gemm_bf16x3(
        const float* __restrict__ X,
        const unsigned short* __restrict__ BhG0,
        const unsigned short* __restrict__ BlG0,
        const float* __restrict__ bias,
        float* __restrict__ out) {
    __shared__ unsigned short sAh[BM][LDSS];
    __shared__ unsigned short sAl[BM][LDSS];
    __shared__ unsigned short sBh[BN][LDSS];
    __shared__ unsigned short sBl[BN][LDSS];

    const int tid   = threadIdx.x;
    const int wave  = tid >> 5;
    const int lane  = tid & 31;
    const int lm    = lane & 15;     // M/N position within 16x16 tile
    const int half  = lane >> 4;     // selects K-half per hardware frag layout
    const int waveM = wave & 3;      // 4 waves along M -> 32 rows each
    const int waveN = wave >> 2;     // 2 waves along N -> 64 cols each

    const size_t rowBase = (size_t)blockIdx.x * BM;
    const size_t colBase = (size_t)blockIdx.y * BN;

    const float*          Xg  = X    + rowBase * KDIM;
    const unsigned short* BhG = BhG0 + colBase * KDIM;
    const unsigned short* BlG = BlG0 + colBase * KDIM;

    // Per-thread staging coordinates over the 128x32 tile (float4/ushort4 granules)
    int rowIdx[4], colIdx[4];
#pragma unroll
    for (int r = 0; r < 4; ++r) {
        int f = tid + r * 256;        // 0..1023
        rowIdx[r] = f >> 3;           // 0..127
        colIdx[r] = (f & 7) * 4;      // 0..28
    }

    float4  av[4];
    ushort4 bhv[4], blv[4];
    auto load_tile = [&](int kt) {
#pragma unroll
        for (int r = 0; r < 4; ++r) {
            const size_t off = (size_t)rowIdx[r] * KDIM + kt + colIdx[r];
            av[r]  = *(const float4*)(Xg + off);
            bhv[r] = *(const ushort4*)(BhG + off);
            blv[r] = *(const ushort4*)(BlG + off);
        }
    };

    v8f acc[2][4] = {};

    load_tile(0);                     // prologue: tile 0 in flight

    for (int kt = 0; kt < KDIM; kt += BK) {
        __syncthreads();              // previous compute finished reading LDS
#pragma unroll
        for (int r = 0; r < 4; ++r) {
            const int row = rowIdx[r], col = colIdx[r];
            ushort4 h, l;
            split1(av[r].x, h.x, l.x);
            split1(av[r].y, h.y, l.y);
            split1(av[r].z, h.z, l.z);
            split1(av[r].w, h.w, l.w);
            *(ushort4*)&sAh[row][col] = h;
            *(ushort4*)&sAl[row][col] = l;
            *(ushort4*)&sBh[row][col] = bhv[r];
            *(ushort4*)&sBl[row][col] = blv[r];
        }
        __syncthreads();

        // Issue next tile's global loads now; their wait lands in the next
        // iteration's store phase, hidden behind this tile's WMMAs.
        if (kt + BK < KDIM) load_tile(kt + BK);

        // ---- fragments: per-lane K is contiguous in two 16B chunks (ds_load_b128 x2)
        v16bf aH[2], aL[2];
#pragma unroll
        for (int t = 0; t < 2; ++t) {
            const unsigned short* p = &sAh[waveM * 32 + t * 16 + lm][half * 8];
            *((v8bf*)&aH[t])     = *(const v8bf*)p;          // K = k0 .. k0+7
            *((v8bf*)&aH[t] + 1) = *(const v8bf*)(p + 16);   // K = k0+16 .. k0+23
            const unsigned short* q = &sAl[waveM * 32 + t * 16 + lm][half * 8];
            *((v8bf*)&aL[t])     = *(const v8bf*)q;
            *((v8bf*)&aL[t] + 1) = *(const v8bf*)(q + 16);
        }
#pragma unroll
        for (int u = 0; u < 4; ++u) {
            v16bf bH, bL;
            const unsigned short* p = &sBh[waveN * 64 + u * 16 + lm][half * 8];
            *((v8bf*)&bH)     = *(const v8bf*)p;
            *((v8bf*)&bH + 1) = *(const v8bf*)(p + 16);
            const unsigned short* q = &sBl[waveN * 64 + u * 16 + lm][half * 8];
            *((v8bf*)&bL)     = *(const v8bf*)q;
            *((v8bf*)&bL + 1) = *(const v8bf*)(q + 16);
#pragma unroll
            for (int t = 0; t < 2; ++t) {
                acc[t][u] = __builtin_amdgcn_wmma_f32_16x16x32_bf16(
                    false, aH[t], false, bH, (short)0, acc[t][u], false, false);
                acc[t][u] = __builtin_amdgcn_wmma_f32_16x16x32_bf16(
                    false, aH[t], false, bL, (short)0, acc[t][u], false, false);
                acc[t][u] = __builtin_amdgcn_wmma_f32_16x16x32_bf16(
                    false, aL[t], false, bH, (short)0, acc[t][u], false, false);
            }
        }
    }

    // ---- epilogue: bias add + store (C/D layout: VGPR r -> M = r + 8*half, N = lm)
#pragma unroll
    for (int u = 0; u < 4; ++u) {
        const size_t n  = colBase + waveN * 64 + u * 16 + lm;
        const float  bv = bias[n];
#pragma unroll
        for (int t = 0; t < 2; ++t) {
            const size_t m0 = rowBase + waveM * 32 + t * 16 + half * 8;
#pragma unroll
            for (int r = 0; r < 8; ++r)
                out[(m0 + r) * (size_t)NDIM + n] = acc[t][u][r] + bv;
        }
    }
}

// ---------------------------------------------------------------------------
extern "C" void kernel_launch(void* const* d_in, const int* in_sizes, int n_in,
                              void* d_out, int out_size, void* d_ws, size_t ws_size,
                              hipStream_t stream) {
    const float* x      = (const float*)d_in[0];   // (8192, 256, 8) f32
    const float* weight = (const float*)d_in[1];   // (256, 256, 8)  f32
    const float* bias   = (const float*)d_in[2];   // (256, 8)       f32
    // d_in[3] = cayley table: signs recomputed on device, not needed.
    float* out = (float*)d_out;                    // (8192, 256, 8) f32

    // Workspace: pre-split effective weight, [n][kk] layout, bf16 hi + bf16 lo.
    // Needs 2 * 2048*2048 * 2 bytes = 16 MB.
    unsigned short* wsBh = (unsigned short*)d_ws;
    unsigned short* wsBl = wsBh + (size_t)NDIM * KDIM;

    clifford_prep_wt<<<(NDIM * KDIM) / 256, 256, 0, stream>>>(weight, wsBh, wsBl);

    dim3 grid(BATCH / BM, NDIM / BN);  // 64 x 16
    clifford_gemm_bf16x3<<<grid, 256, 0, stream>>>(x, wsBh, wsBl, bias, out);

    (void)in_sizes; (void)n_in; (void)out_size; (void)ws_size;
}
